// RelLSTM_45440753992142
// MI455X (gfx1250) — matrix operand order repaired and checked by reading, not compiled, not run on hardware
//
#include <hip/hip_runtime.h>

// Problem constants (must match reference)
#define T_STEPS  512
#define NB       64
#define NI       128
#define NH       256
#define LASTK    16
#define RSZ      32
#define NTOT     48            // LASTK + RSZ
#define KCAT     384           // NH + NI
#define NGATE    1024          // 4 * NH
#define BH       (NB * NH)     // 16384
#define NTHREADS 1024
#define NWAVES   32

typedef __attribute__((ext_vector_type(16))) _Float16 v16h;
typedef __attribute__((ext_vector_type(8)))  _Float16 v8h;
typedef __attribute__((ext_vector_type(4)))  _Float16 v4h;
typedef __attribute__((ext_vector_type(8)))  float    v8f;
typedef __attribute__((ext_vector_type(4)))  float    v4f;

// ---------------- workspace layout (byte offsets, all 16B aligned) --------
constexpr size_t OFF_WT4   = 0;                                        // f16 packed frags [KCAT x NGATE]
constexpr size_t OFF_UAT   = OFF_WT4   + (size_t)KCAT * NGATE * 2;     // f16 packed frags (Ua^T)
constexpr size_t OFF_VAT   = OFF_UAT   + (size_t)NH * NH * 2;          // f16 packed frags (Va^T)
constexpr size_t OFF_MEM   = OFF_VAT   + (size_t)NH * NH * 2;          // f32 [LASTK][NB][NH] ring
constexpr size_t OFF_UAH   = OFF_MEM   + (size_t)LASTK * BH * 4;       // f32 cached Ua@mem ring
constexpr size_t OFF_LMEM  = OFF_UAH   + (size_t)LASTK * BH * 4;       // f32 [RSZ][NB][NH]
constexpr size_t OFF_LUA   = OFF_LMEM  + (size_t)RSZ * BH * 4;         // f32 cached Ua@long_mem
constexpr size_t OFF_LMASK = OFF_LUA   + (size_t)RSZ * BH * 4;         // f32 [RSZ][NB]
constexpr size_t OFF_LSC   = OFF_LMASK + (size_t)RSZ * NB * 4;         // f32 [T][NB]
constexpr size_t OFF_BUCK  = OFF_LSC   + (size_t)T_STEPS * NB * 4;     // f32 [NB][RSZ]
constexpr size_t OFF_FILL  = OFF_BUCK  + (size_t)NB * RSZ * 4;         // i32 [NB]
constexpr size_t OFF_ST    = OFF_FILL  + (size_t)NB * 4;               // f32 [NB][NH]
constexpr size_t OFF_CT    = OFF_ST    + (size_t)BH * 4;               // f32 [NB][NH]
constexpr size_t OFF_Q     = OFF_CT    + (size_t)BH * 4;               // f32 [NB][NH]
constexpr size_t OFF_GATES = OFF_Q     + (size_t)BH * 4;               // f32 [NB][NGATE]
constexpr size_t OFF_HBUF  = OFF_GATES + (size_t)NB * NGATE * 4;       // f32 [NB][NH]
constexpr size_t OFF_INP16 = OFF_HBUF  + (size_t)BH * 4;               // f16 [NB][KCAT]
constexpr size_t OFF_H16   = OFF_INP16 + (size_t)NB * KCAT * 2;        // f16 [NB][NH]

__device__ __forceinline__ float sigmf(float x) { return 1.0f / (1.0f + __expf(-x)); }

// ---------------------------------------------------------------------------
// Cooperative f16 WMMA GEMM: C[M][N](f32) = A[M][K](f16,row-major) * Bp,
// where Bp holds the B matrix pre-swizzled in per-lane fragment order:
//   Bp[(((kt * (N/16)) + tn) * 32 + lane) * 16 + i] = B[kt*32 + (lane>>4)*16 + i][tn*16 + (lane&15)]
// so each lane's 16 halfs for a WMMA are contiguous (two 16B loads).
// All 32 waves participate; tile counts are multiples of 32 so control flow
// is wave-uniform and EXEC stays all-ones (WMMA requirement).
// ---------------------------------------------------------------------------
static __device__ void gemm_f16_wg(const _Float16* __restrict__ A, int lda,
                                   const _Float16* __restrict__ Bp,
                                   float* __restrict__ C, int ldc,
                                   int M, int N, int K)
{
    const int lane = threadIdx.x & 31;
    const int wave = threadIdx.x >> 5;
    const int half = lane >> 4;     // 0 or 1
    const int l15  = lane & 15;
    const int mt = M >> 4, nt = N >> 4, kt = K >> 5;
    const int ntiles = mt * nt;
    const size_t bstride = (size_t)nt * 512;   // halfs per k-tile row of fragments

    for (int tile = wave; tile < ntiles; tile += NWAVES) {
        const int tmi = tile / nt, tni = tile % nt;
        const int tm = tmi << 4;
        const _Float16* arow  = A  + (size_t)(tm + l15) * lda + half * 16;
        const _Float16* bbase = Bp + ((size_t)tni * 32 + lane) * 16;
        v8f acc = {};
        for (int kk = 0; kk < kt; ++kk) {
            union { v16h v; v8h h[2]; } a, b;
            const _Float16* ap = arow + kk * 32;
            a.h[0] = *(const v8h*)(ap);
            a.h[1] = *(const v8h*)(ap + 8);
            const _Float16* bp = bbase + kk * bstride;
            b.h[0] = *(const v8h*)(bp);
            b.h[1] = *(const v8h*)(bp + 8);
            acc = __builtin_amdgcn_wmma_f32_16x16x32_f16(
                false, a.v, false, b.v, (short)0, acc, false, false);
        }
        float* cp = C + (size_t)(tm + 8 * half) * ldc + (tni << 4) + l15;
#pragma unroll
        for (int r = 0; r < 8; ++r) cp[(size_t)r * ldc] = acc[r];
    }
}

// ---------------------------------------------------------------------------
// Persistent single-workgroup kernel: runs the whole T=512 scan.
// ---------------------------------------------------------------------------
__global__ __launch_bounds__(NTHREADS)
void rel_lstm_persistent(const float* __restrict__ x,
                         const float* __restrict__ Wf_w, const float* __restrict__ Wf_b,
                         const float* __restrict__ Wi_w, const float* __restrict__ Wi_b,
                         const float* __restrict__ Wo_w, const float* __restrict__ Wo_b,
                         const float* __restrict__ Wg_w, const float* __restrict__ Wg_b,
                         const float* __restrict__ Ua_w, const float* __restrict__ Va_w,
                         const float* __restrict__ vvec,
                         float* __restrict__ out, char* __restrict__ ws)
{
    const int tid = threadIdx.x;

    _Float16* wt4   = (_Float16*)(ws + OFF_WT4);
    _Float16* uat   = (_Float16*)(ws + OFF_UAT);
    _Float16* vat   = (_Float16*)(ws + OFF_VAT);
    float*    mem   = (float*)(ws + OFF_MEM);
    float*    uah   = (float*)(ws + OFF_UAH);
    float*    lmem  = (float*)(ws + OFF_LMEM);
    float*    lua   = (float*)(ws + OFF_LUA);
    float*    lmask = (float*)(ws + OFF_LMASK);
    float*    lsc   = (float*)(ws + OFF_LSC);
    float*    buck  = (float*)(ws + OFF_BUCK);
    int*      fill  = (int*)(ws + OFF_FILL);
    float*    st    = (float*)(ws + OFF_ST);
    float*    ct    = (float*)(ws + OFF_CT);
    float*    q     = (float*)(ws + OFF_Q);
    float*    gates = (float*)(ws + OFF_GATES);
    float*    hbuf  = (float*)(ws + OFF_HBUF);
    _Float16* inp16 = (_Float16*)(ws + OFF_INP16);
    _Float16* h16   = (_Float16*)(ws + OFF_H16);

    __shared__ float es_s[NTOT][NB];   // attention logits -> alphas
    __shared__ float vv_s[NH];
    __shared__ float bias_s[4][NH];
    __shared__ int   addpos_s[NB];
    __shared__ int   upd_s[NB];

    const int lane = tid & 31;
    const int wave = tid >> 5;
    const float NEG_INF = -__builtin_inff();

    // ================= init: weight conversion + state reset ===============
    // wt4: packed fragments of B[k][n] = W_{n>>8}[n&255][k], N = 1024
    for (int idx = tid; idx < KCAT * NGATE; idx += NTHREADS) {
        const int i    = idx & 15;
        const int ln   = (idx >> 4) & 31;
        const int rest = idx >> 9;
        const int tn   = rest % (NGATE >> 4);
        const int ktb  = rest / (NGATE >> 4);
        const int k = (ktb << 5) + ((ln >> 4) << 4) + i;
        const int n = (tn << 4) + (ln & 15);
        const int g = n >> 8, h = n & 255;
        const float* W = (g == 0) ? Wf_w : (g == 1) ? Wi_w : (g == 2) ? Wo_w : Wg_w;
        wt4[idx] = (_Float16)W[(size_t)h * KCAT + k];
    }
    // uat/vat: packed fragments of B[k][n] = Ua_w[n][k] (resp. Va_w), N = 256
    for (int idx = tid; idx < NH * NH; idx += NTHREADS) {
        const int i    = idx & 15;
        const int ln   = (idx >> 4) & 31;
        const int rest = idx >> 9;
        const int tn   = rest & 15;          // N/16 == 16
        const int ktb  = rest >> 4;
        const int k = (ktb << 5) + ((ln >> 4) << 4) + i;
        const int n = (tn << 4) + (ln & 15);
        uat[idx] = (_Float16)Ua_w[(size_t)n * NH + k];
        vat[idx] = (_Float16)Va_w[(size_t)n * NH + k];
    }
    for (int idx = tid; idx < LASTK * BH; idx += NTHREADS) { mem[idx] = 0.0f; uah[idx] = 0.0f; }
    for (int idx = tid; idx < RSZ * BH;   idx += NTHREADS) { lmem[idx] = 0.0f; lua[idx] = 0.0f; }
    for (int idx = tid; idx < RSZ * NB;   idx += NTHREADS) lmask[idx] = NEG_INF;
    for (int idx = tid; idx < T_STEPS * NB; idx += NTHREADS) lsc[idx] = 0.0f;
    for (int idx = tid; idx < NB * RSZ;   idx += NTHREADS) buck[idx] = 0.0f;
    for (int idx = tid; idx < BH;         idx += NTHREADS) { st[idx] = 0.0f; ct[idx] = 0.0f; }
    if (tid < NB) fill[tid] = 0;
    if (tid < NH) {
        vv_s[tid] = vvec[tid];
        bias_s[0][tid] = Wf_b[tid];
        bias_s[1][tid] = Wi_b[tid];
        bias_s[2][tid] = Wo_b[tid];
        bias_s[3][tid] = Wg_b[tid];
    }
    __syncthreads();

    // ================= t = 0 (reset path, no attention) ====================
    for (int idx = tid * 4; idx < NB * KCAT; idx += NTHREADS * 4) {
        const int b = idx / KCAT, k = idx % KCAT;
        v4f v = {};
        if (k >= NH) v = *(const v4f*)(x + (size_t)b * NI + (k - NH));
        *(v4h*)(inp16 + idx) = __builtin_convertvector(v, v4h);
    }
    __syncthreads();
    gemm_f16_wg(inp16, KCAT, wt4, gates, NGATE, NB, NGATE, KCAT);
    __syncthreads();
    for (int e = tid; e < BH; e += NTHREADS) {
        const int b = e >> 8, h = e & 255;
        const float* gr = gates + (size_t)b * NGATE;
        const float ff = sigmf(gr[h] + bias_s[0][h]);
        const float ii = sigmf(gr[NH + h] + bias_s[1][h]);
        const float oo = sigmf(gr[2 * NH + h] + bias_s[2][h]);
        const float gg = tanhf(gr[3 * NH + h] + bias_s[3][h]);
        (void)ff;
        const float cv = ii * gg;                 // ct0 = 0
        const float hid = oo * tanhf(cv);
        ct[e] = cv;
        out[e] = hid;                             // out[0]
        mem[(size_t)15 * BH + e] = hid;           // logical slot LASTK-1 @ t=1
        h16[e] = (_Float16)hid;
    }
    __syncthreads();
    gemm_f16_wg(h16, NH, uat, uah + (size_t)15 * BH, NH, NB, NH, NH);
    __syncthreads();

    // ================= scan body: t = 1 .. T-1 =============================
    for (int t = 1; t < T_STEPS; ++t) {
        const int p0    = (t - 1) & 15;        // phys of logical slot 0 (h_{t-16})
        const int pprev = (t + 14) & 15;       // phys of logical slot 15 (h_{t-1})

        // ---- q = st @ Va^T (st is the carried attention state) ----
        for (int e0 = tid * 4; e0 < BH; e0 += NTHREADS * 4)
            *(v4h*)(h16 + e0) = __builtin_convertvector(*(const v4f*)(st + e0), v4h);
        if (tid == 0 && t + 1 < T_STEPS)
            __builtin_prefetch(x + (size_t)(t + 1) * NB * NI, 0, 1);
        __syncthreads();
        gemm_f16_wg(h16, NH, vat, q, NH, NB, NH, NH);
        __syncthreads();

        // ---- attention logits over 16 short slots + 32 buckets ----
        // lane owns a contiguous 8-wide chunk of H: two v4f loads per operand.
        for (int p = wave; p < NTOT * NB; p += NWAVES) {
            const int n = p / NB, b = p % NB;
            const float* U = (n < LASTK)
                ? (uah + (size_t)((t - 1 + n) & 15) * BH + (size_t)b * NH)
                : (lua + (size_t)(n - LASTK) * BH + (size_t)b * NH);
            const float* qb = q + (size_t)b * NH;
            const int hb = lane * 8;
            const v4f q0 = *(const v4f*)(qb + hb), q1 = *(const v4f*)(qb + hb + 4);
            const v4f u0 = *(const v4f*)(U + hb),  u1 = *(const v4f*)(U + hb + 4);
            float part = vv_s[hb + 0] * tanhf(q0.x + u0.x)
                       + vv_s[hb + 1] * tanhf(q0.y + u0.y)
                       + vv_s[hb + 2] * tanhf(q0.z + u0.z)
                       + vv_s[hb + 3] * tanhf(q0.w + u0.w)
                       + vv_s[hb + 4] * tanhf(q1.x + u1.x)
                       + vv_s[hb + 5] * tanhf(q1.y + u1.y)
                       + vv_s[hb + 6] * tanhf(q1.z + u1.z)
                       + vv_s[hb + 7] * tanhf(q1.w + u1.w);
            for (int off = 16; off > 0; off >>= 1)
                part += __shfl_xor(part, off, 32);
            float e;
            if (n < LASTK) e = (n >= LASTK - t) ? part : NEG_INF;
            else           e = part + lmask[(size_t)(n - LASTK) * NB + b];
            if (lane == 0) es_s[n][b] = e;
        }
        __syncthreads();

        // ---- softmax over the 48 entries (one thread per batch sample) ----
        if (tid < NB) {
            float mx = NEG_INF;
            for (int n = 0; n < NTOT; ++n) mx = fmaxf(mx, es_s[n][tid]);
            float s = 0.0f;
            for (int n = 0; n < NTOT; ++n) {
                const float a = __expf(es_s[n][tid] - mx);
                es_s[n][tid] = a; s += a;
            }
            const float inv = 1.0f / s;
            for (int n = 0; n < NTOT; ++n) es_s[n][tid] *= inv;
        }
        __syncthreads();

        // ---- accumulate detached short-slot mass onto absolute-time scores
        {
            const int j = tid >> 6, b = tid & 63;  // 16 x 64 pairs
            if (j >= LASTK - t)
                lsc[(size_t)(t - LASTK + j) * NB + b] += es_s[j][b];
        }

        // ---- ct_att and new attention state st (float4-vectorized) ----
        for (int e0 = tid * 4; e0 < BH; e0 += NTHREADS * 4) {
            const int b = e0 >> 8;
            v4f acc = {};
#pragma unroll
            for (int n = 0; n < LASTK; ++n) {
                const v4f m4 = *(const v4f*)(mem + (size_t)((t - 1 + n) & 15) * BH + e0);
                acc += es_s[n][b] * m4;
            }
#pragma unroll
            for (int r = 0; r < RSZ; ++r) {
                const v4f m4 = *(const v4f*)(lmem + (size_t)r * BH + e0);
                acc += es_s[LASTK + r][b] * m4;
            }
            const v4f prev = *(const v4f*)(mem + (size_t)pprev * BH + e0);
            *(v4f*)(st + e0) = 0.5f * (prev + acc);
        }
        __syncthreads();

        // ---- LSTM gates: [st | x_t] @ W^T for all 4 gates at once ----
        for (int idx = tid * 4; idx < NB * KCAT; idx += NTHREADS * 4) {
            const int b = idx / KCAT, k = idx % KCAT;
            const v4f v = (k < NH)
                ? *(const v4f*)(st + (size_t)b * NH + k)
                : *(const v4f*)(x + ((size_t)t * NB + b) * NI + (k - NH));
            *(v4h*)(inp16 + idx) = __builtin_convertvector(v, v4h);
        }
        __syncthreads();
        gemm_f16_wg(inp16, KCAT, wt4, gates, NGATE, NB, NGATE, KCAT);
        __syncthreads();
        for (int e = tid; e < BH; e += NTHREADS) {
            const int b = e >> 8, h = e & 255;
            const float* gr = gates + (size_t)b * NGATE;
            const float ff = sigmf(gr[h] + bias_s[0][h]);
            const float ii = sigmf(gr[NH + h] + bias_s[1][h]);
            const float oo = sigmf(gr[2 * NH + h] + bias_s[2][h]);
            const float gg = tanhf(gr[3 * NH + h] + bias_s[3][h]);
            const float cv = ff * ct[e] + ii * gg;
            const float hid = oo * tanhf(cv);
            ct[e] = cv;
            out[(size_t)t * BH + e] = hid;
            hbuf[e] = hid;
            h16[e] = (_Float16)hid;
        }
        __syncthreads();

        // ---- bucket insert (argmin-replace, active once t >= LASTK) ----
        if (tid < NB) {
            const int b = tid;
            const int fl = fill[b];
            float mn = buck[(size_t)b * RSZ]; int mp = 0;
            for (int r = 1; r < RSZ; ++r) {
                const float bv = buck[(size_t)b * RSZ + r];
                if (bv < mn) { mn = bv; mp = r; }      // first-min
            }
            const bool full = fl >= RSZ;
            const int  ap = full ? mp : fl;
            const int  tc = (t - LASTK) > 0 ? (t - LASTK) : 0;
            const float cs = lsc[(size_t)tc * NB + b];
            const bool du = (t >= LASTK) && (!full || (cs > mn));
            addpos_s[b] = ap; upd_s[b] = du ? 1 : 0;
            if (du) {
                buck[(size_t)b * RSZ + ap] = cs;
                lmask[(size_t)ap * NB + b] = 0.0f;
                if (!full) fill[b] = fl + 1;
            }
        }
        __syncthreads();

        // ---- promote cand (h_{t-16}, cached Ua@h) then push h_t in ring ----
        for (int e0 = tid * 4; e0 < BH; e0 += NTHREADS * 4) {
            const int b = e0 >> 8;
            const v4f cand_v = *(const v4f*)(mem + (size_t)p0 * BH + e0);
            const v4f cand_u = *(const v4f*)(uah + (size_t)p0 * BH + e0);
            if (upd_s[b]) {
                const int ap = addpos_s[b];
                *(v4f*)(lmem + (size_t)ap * BH + e0) = cand_v;
                *(v4f*)(lua + (size_t)ap * BH + e0)  = cand_u;
            }
            *(v4f*)(mem + (size_t)p0 * BH + e0) = *(const v4f*)(hbuf + e0);
        }
        __syncthreads();
        gemm_f16_wg(h16, NH, uat, uah + (size_t)p0 * BH, NH, NB, NH, NH);
        __syncthreads();
    }
}

// ---------------------------------------------------------------------------
extern "C" void kernel_launch(void* const* d_in, const int* in_sizes, int n_in,
                              void* d_out, int out_size, void* d_ws, size_t ws_size,
                              hipStream_t stream) {
    (void)in_sizes; (void)n_in; (void)out_size; (void)ws_size;
    const float* x    = (const float*)d_in[0];
    const float* Wf_w = (const float*)d_in[1];
    const float* Wf_b = (const float*)d_in[2];
    const float* Wi_w = (const float*)d_in[3];
    const float* Wi_b = (const float*)d_in[4];
    const float* Wo_w = (const float*)d_in[5];
    const float* Wo_b = (const float*)d_in[6];
    const float* Wg_w = (const float*)d_in[7];
    const float* Wg_b = (const float*)d_in[8];
    const float* Ua_w = (const float*)d_in[9];
    const float* Va_w = (const float*)d_in[10];
    const float* vv   = (const float*)d_in[11];

    rel_lstm_persistent<<<1, NTHREADS, 0, stream>>>(
        x, Wf_w, Wf_b, Wi_w, Wi_b, Wo_w, Wo_b, Wg_w, Wg_b,
        Ua_w, Va_w, vv, (float*)d_out, (char*)d_ws);
}
